// GRU_64484638982316
// MI455X (gfx1250) — matrix operand order repaired and checked by reading, not compiled, not run on hardware
//
#include <hip/hip_runtime.h>

typedef __attribute__((ext_vector_type(16))) _Float16 v16h;
typedef __attribute__((ext_vector_type(8)))  _Float16 v8h;
typedef __attribute__((ext_vector_type(8)))  float    v8f;

#define B_   32
#define C_   64
#define H_   64
#define L_   32768
#define CH_  128   // C_ + H_

// LDS budget:
//   sW:    48 frags * 32 lanes * 16 f16 * 2B = 49152 B  (A-fragments, pre-swizzled)
//   sBias: 3 * 64 f32                        =   768 B
//   sHr:   8 waves * 16 cols * 40 rows * 2B  = 10240 B  (C-layout -> B-layout bounce)
//   total                                    = 60160 B

__device__ __forceinline__ v16h ldFragA(const _Float16* sW, int f, int lane) {
    // [frag][lane][16 f16], 32B per lane -> two ds_load_b128, conflict-free
    return *(const v16h*)&sW[(f * 32 + lane) * 16];
}

__device__ __forceinline__ v8f ldBias(const float* sBias, int g, int mt, int half) {
    const float4 b0 = *(const float4*)&sBias[g * 64 + mt * 16 + half * 8];
    const float4 b1 = *(const float4*)&sBias[g * 64 + mt * 16 + half * 8 + 4];
    v8f r;
    r[0] = b0.x; r[1] = b0.y; r[2] = b0.z; r[3] = b0.w;
    r[4] = b1.x; r[5] = b1.y; r[6] = b1.z; r[7] = b1.w;
    return r;
}

__global__ __launch_bounds__(256)
void gru_wmma_kernel(const float* __restrict__ x,
                     const float* __restrict__ h,
                     const float* __restrict__ Wr, const float* __restrict__ br,
                     const float* __restrict__ Wu, const float* __restrict__ bu,
                     const float* __restrict__ Wh, const float* __restrict__ bh,
                     float* __restrict__ out)
{
    __shared__ __align__(16) _Float16 sW[48 * 32 * 16];   // [frag][lane][16 f16]
    __shared__ __align__(16) float    sBias[3 * 64];
    __shared__ __align__(16) _Float16 sHr[8 * 16 * 40];   // [wave][col][row pad40]

    const int tid  = threadIdx.x;
    const int wave = tid >> 5;
    const int lane = tid & 31;
    const int n    = lane & 15;   // N / column index inside a 16-tile
    const int half = lane >> 4;   // lane half (K/M split per ISA layouts)

    // ---- stage W_{r,u,h} as f16 WMMA A-fragments in LDS ----
    // A 16x32 f16 layout: m = mt*16 + lane%16 ; K(i) = kt*32 + half*16 + i
    for (int f = wave; f < 48; f += 8) {            // f = g*16 + kt*4 + mt
        const int g  = f >> 4;
        const int kt = (f >> 2) & 3;
        const int mt = f & 3;
        const float* Wg = (g == 0) ? Wr : (g == 1) ? Wu : Wh;
        const int m  = mt * 16 + n;
        const int kb = kt * 32 + half * 16;
        _Float16* dst = &sW[(f * 32 + lane) * 16];
#pragma unroll
        for (int i = 0; i < 8; ++i) dst[i]     = (_Float16)Wg[m * CH_ + kb + i];
#pragma unroll
        for (int i = 0; i < 8; ++i) dst[8 + i] = (_Float16)Wg[m * CH_ + kb + 8 + i];
    }
    for (int t = tid; t < 192; t += 256) {
        const float* bg = (t < 64) ? br : (t < 128) ? bu : bh;
        sBias[t] = bg[t & 63];
    }
    __syncthreads();

    const int    b     = blockIdx.x;
    const size_t xBase = (size_t)b * C_ * L_;
    const size_t hBase = (size_t)b * H_ * L_;
    _Float16*    myHr  = &sHr[wave * 16 * 40];

#pragma unroll 1
    for (int it = 0; it < 4; ++it) {
        // Prevent hoisting of the (loop-invariant) LDS fragment/bias reads:
        // without this the compiler preloads all 48 A-frags -> ~500 VGPRs + spills.
        asm volatile("" ::: "memory");

        const int tile = blockIdx.y * 32 + it * 8 + wave;  // 2048 tiles of 16 along L
        const size_t col = (size_t)(tile * 16 + n);

        // ---- x, h as B-fragments (32xN16 f16): K(i) = kt*32 + half*16 + i ----
        v16h xb[2], hb[2];
#pragma unroll
        for (int kt = 0; kt < 2; ++kt) {
#pragma unroll
            for (int i = 0; i < 16; ++i) {
                const int c = kt * 32 + half * 16 + i;
                xb[kt][i] = (_Float16)x[xBase + (size_t)c * L_ + col];
                hb[kt][i] = (_Float16)h[hBase + (size_t)c * L_ + col];
            }
        }
        // ---- h in f32 C-layout (row m = mt*16 + half*8 + v, col n) ----
        float hc[4][8];
#pragma unroll
        for (int mt = 0; mt < 4; ++mt)
#pragma unroll
            for (int v = 0; v < 8; ++v)
                hc[mt][v] = h[hBase + (size_t)(mt * 16 + half * 8 + v) * L_ + col];

        // ---- gates r, u : 4 Mtiles x 4 Ktiles each ----
        float R[4][8];
        v8h   uPk[4];     // u kept packed f16 to cut register pressure
#pragma unroll
        for (int mt = 0; mt < 4; ++mt) {
            v8f ar = ldBias(sBias, 0, mt, half);
            v8f au = ldBias(sBias, 1, mt, half);
#pragma unroll
            for (int kt = 0; kt < 4; ++kt) {
                const v16h bf = (kt < 2) ? xb[kt] : hb[kt - 2];
                const v16h aR = ldFragA(sW, 0 * 16 + kt * 4 + mt, lane);
                const v16h aU = ldFragA(sW, 1 * 16 + kt * 4 + mt, lane);
                ar = __builtin_amdgcn_wmma_f32_16x16x32_f16(false, aR, false, bf, (short)0, ar, false, false);
                au = __builtin_amdgcn_wmma_f32_16x16x32_f16(false, aU, false, bf, (short)0, au, false, false);
            }
#pragma unroll
            for (int v = 0; v < 8; ++v) {
                R[mt][v]   = 1.0f / (1.0f + __expf(-ar[v]));
                uPk[mt][v] = (_Float16)(1.0f / (1.0f + __expf(-au[v])));
            }
        }

        // ---- gate n : x part first ----
        v8f accN[4];
#pragma unroll
        for (int mt = 0; mt < 4; ++mt) {
            v8f an = ldBias(sBias, 2, mt, half);
#pragma unroll
            for (int kt = 0; kt < 2; ++kt) {
                const v16h aN = ldFragA(sW, 2 * 16 + kt * 4 + mt, lane);
                an = __builtin_amdgcn_wmma_f32_16x16x32_f16(false, aN, false, xb[kt], (short)0, an, false, false);
            }
            accN[mt] = an;
        }
        // ---- (h * r) part: C-layout -> LDS -> B-fragment, two 32-row passes ----
#pragma unroll
        for (int p = 0; p < 2; ++p) {
#pragma unroll
            for (int q = 0; q < 2; ++q) {      // C-layout Mtile = p*2 + q
                const int mt = p * 2 + q;
                v8h t;
#pragma unroll
                for (int v = 0; v < 8; ++v) t[v] = (_Float16)(hc[mt][v] * R[mt][v]);
                *(v8h*)&myHr[n * 40 + q * 16 + half * 8] = t;   // 16B aligned
            }
            // B-frag for K-tile (2+p): element i = hr row half*16 + i, col n
            const v16h hrb = *(const v16h*)&myHr[n * 40 + half * 16];
#pragma unroll
            for (int mt = 0; mt < 4; ++mt) {
                const v16h aN = ldFragA(sW, 2 * 16 + (2 + p) * 4 + mt, lane);
                accN[mt] = __builtin_amdgcn_wmma_f32_16x16x32_f16(false, aN, false, hrb, (short)0, accN[mt], false, false);
            }
        }

        // ---- tanh + blend + store (coalesced 64B rows) ----
#pragma unroll
        for (int mt = 0; mt < 4; ++mt) {
#pragma unroll
            for (int v = 0; v < 8; ++v) {
                const float e  = __expf(-2.0f * accN[mt][v]);
                const float nn = (1.0f - e) / (1.0f + e);          // tanh
                const float u  = (float)uPk[mt][v];
                out[hBase + (size_t)(mt * 16 + half * 8 + v) * L_ + col]
                    = nn + u * (hc[mt][v] - nn);                   // (1-u)*n + u*h
            }
        }
    }
}

extern "C" void kernel_launch(void* const* d_in, const int* in_sizes, int n_in,
                              void* d_out, int out_size, void* d_ws, size_t ws_size,
                              hipStream_t stream) {
    (void)in_sizes; (void)n_in; (void)out_size; (void)d_ws; (void)ws_size;
    const float* x  = (const float*)d_in[0];
    const float* h  = (const float*)d_in[1];
    const float* Wr = (const float*)d_in[2];
    const float* br = (const float*)d_in[3];
    const float* Wu = (const float*)d_in[4];
    const float* bu = (const float*)d_in[5];
    const float* Wh = (const float*)d_in[6];
    const float* bh = (const float*)d_in[7];
    float* out = (float*)d_out;

    dim3 grid(B_, 64);   // 32 batches x 64 L-chunks; 8 waves/block, 4 tiles/wave
    gru_wmma_kernel<<<grid, 256, 0, stream>>>(x, h, Wr, br, Wu, bu, Wh, bh, out);
}
